// TRADE_738734374996
// MI455X (gfx1250) — compile-verified, matrix operand order checked by reading, not run on hardware
//
#include <hip/hip_runtime.h>
#include <hip/hip_bf16.h>
#include <math.h>

#define V_    18000
#define H_    400
#define B_    16
#define TENC_ 256
#define S_    30
#define TDEC_ 8
#define G_    3
#define KP_   416          // K padded to multiple of 32 (13 * 32)
#define SB_   (S_ * B_)    // 480
#define H3_   (3 * H_)     // 1200

typedef __bf16 bf16;
typedef __attribute__((ext_vector_type(16))) __bf16 v16bf;
typedef __attribute__((ext_vector_type(8)))  float  v8f;

static __device__ __forceinline__ float sigmoidf_(float x) {
    return 1.f / (1.f + __expf(-x));
}

// ---------------------------------------------------------------------------
// Convert f32 (rows x 400) -> bf16 (rows x 416), zero padded K tail.
// ---------------------------------------------------------------------------
__global__ void k_pad_bf16(const float* __restrict__ src, bf16* __restrict__ dst,
                           int rows, int cols) {
    int total = rows * KP_;
    for (int i = blockIdx.x * blockDim.x + threadIdx.x; i < total;
         i += gridDim.x * blockDim.x) {
        int r = i / KP_, c = i % KP_;
        dst[i] = (c < cols) ? (bf16)src[(size_t)r * cols + c] : (bf16)0.f;
    }
}

// ---------------------------------------------------------------------------
// Gather story embeddings into x_bf16 laid out (t*B + b) x KP_, zero-padded.
// ---------------------------------------------------------------------------
__global__ void k_embed_story(const int* __restrict__ story,
                              const float* __restrict__ emb,
                              bf16* __restrict__ xbf) {
    int total = TENC_ * B_ * KP_;
    for (int i = blockIdx.x * blockDim.x + threadIdx.x; i < total;
         i += gridDim.x * blockDim.x) {
        int r = i / KP_, c = i % KP_;
        int t = r / B_, b = r % B_;
        int tok = story[b * TENC_ + t];
        xbf[i] = (c < H_) ? (bf16)emb[(size_t)tok * H_ + c] : (bf16)0.f;
    }
}

// ---------------------------------------------------------------------------
// Generic C = A * B^T via v_wmma_f32_16x16x32_bf16.
//   A: M x KP_ bf16 row-major, B: N x KP_ bf16 row-major, C: M x N f32.
// One wave computes a 16(M) x 48(N) tile (3 accumulators reuse the A frag).
// Requires M % 16 == 0, N % 48 == 0.
// ---------------------------------------------------------------------------
__global__ __launch_bounds__(256)
void k_gemm_nt(const bf16* __restrict__ A, const bf16* __restrict__ Bm,
               float* __restrict__ C, int M, int N) {
    const int lane = threadIdx.x & 31;
    const int wave = threadIdx.x >> 5;
    const int tile = blockIdx.x * (blockDim.x >> 5) + wave;
    const int ntn  = N / 48;
    const int tm   = tile / ntn;
    const int tn   = tile % ntn;
    if (tm >= M / 16) return;
    const int m0   = tm * 16, n0 = tn * 48;
    const int arow = lane & 15;
    const int koff = (lane >> 4) * 16;

    v8f acc0 = {}, acc1 = {}, acc2 = {};
    for (int k0 = 0; k0 < KP_; k0 += 32) {
        v16bf a  = *(const v16bf*)(A  + (size_t)(m0 + arow)      * KP_ + k0 + koff);
        v16bf b0 = *(const v16bf*)(Bm + (size_t)(n0 + arow)      * KP_ + k0 + koff);
        v16bf b1 = *(const v16bf*)(Bm + (size_t)(n0 + 16 + arow) * KP_ + k0 + koff);
        v16bf b2 = *(const v16bf*)(Bm + (size_t)(n0 + 32 + arow) * KP_ + k0 + koff);
        if (k0 + 32 < KP_)
            __builtin_prefetch(Bm + (size_t)(n0 + arow) * KP_ + k0 + 32 + koff, 0, 0);
        acc0 = __builtin_amdgcn_wmma_f32_16x16x32_bf16(false, a, false, b0,
                                                       (short)0, acc0, false, false);
        acc1 = __builtin_amdgcn_wmma_f32_16x16x32_bf16(false, a, false, b1,
                                                       (short)0, acc1, false, false);
        acc2 = __builtin_amdgcn_wmma_f32_16x16x32_bf16(false, a, false, b2,
                                                       (short)0, acc2, false, false);
    }
    const int crow = (lane >> 4) * 8;
    const int ccol = lane & 15;
    for (int r = 0; r < 8; ++r) {
        C[(size_t)(m0 + crow + r) * N + n0 +      ccol] = acc0[r];
        C[(size_t)(m0 + crow + r) * N + n0 + 16 + ccol] = acc1[r];
        C[(size_t)(m0 + crow + r) * N + n0 + 32 + ccol] = acc2[r];
    }
}

// ---------------------------------------------------------------------------
// Persistent bidirectional GRU encoder recurrence. grid = 2 (dir), 800 thr =
// 25 waves: each wave owns one 16x48 WMMA tile of gh = h_bf16 @ Whh^T
// (16 x 1200). h is kept in LDS (f32 + bf16 fragments); gi precomputed.
// ---------------------------------------------------------------------------
__global__ __launch_bounds__(800)
void k_enc_gru(const bf16* __restrict__ Whh_fp, const bf16* __restrict__ Whh_bp,
               const float* __restrict__ gi_fp, const float* __restrict__ gi_bp,
               const float* __restrict__ bih_fp, const float* __restrict__ bhh_fp,
               const float* __restrict__ bih_bp, const float* __restrict__ bhh_bp,
               float* __restrict__ out_f, float* __restrict__ out_b,
               float* __restrict__ h_f, float* __restrict__ h_b,
               float* __restrict__ gh_ws) {
    const int dir = blockIdx.x;
    const bf16*  Whh  = dir ? Whh_bp : Whh_fp;
    const float* gi   = dir ? gi_bp  : gi_fp;
    const float* bih  = dir ? bih_bp : bih_fp;
    const float* bhh  = dir ? bhh_bp : bhh_fp;
    float* outp = dir ? out_b : out_f;
    float* hfin = dir ? h_b   : h_f;
    float* gh   = gh_ws + (size_t)dir * 16 * H3_;

    __shared__ bf16  hbf[16 * KP_];
    __shared__ float hcur[16 * H_];
    for (int i = threadIdx.x; i < 16 * KP_; i += blockDim.x) hbf[i] = (bf16)0.f;
    for (int i = threadIdx.x; i < 16 * H_;  i += blockDim.x) hcur[i] = 0.f;
    __syncthreads();

    const int lane = threadIdx.x & 31;
    const int wave = threadIdx.x >> 5;      // 0..24 -> N tile
    const int arow = lane & 15;
    const int koff = (lane >> 4) * 16;
    const int crow = (lane >> 4) * 8;
    const int ccol = lane & 15;
    const int n0   = wave * 48;

    for (int t = 0; t < TENC_; ++t) {
        const int tx = dir ? (TENC_ - 1 - t) : t;
        // --- gh tile: 16x48 slice of h @ Whh^T, A fragments from LDS ---
        v8f acc0 = {}, acc1 = {}, acc2 = {};
        for (int k0 = 0; k0 < KP_; k0 += 32) {
            v16bf a  = *(const v16bf*)(&hbf[arow * KP_ + k0 + koff]);
            v16bf b0 = *(const v16bf*)(Whh + (size_t)(n0 + arow)      * KP_ + k0 + koff);
            v16bf b1 = *(const v16bf*)(Whh + (size_t)(n0 + 16 + arow) * KP_ + k0 + koff);
            v16bf b2 = *(const v16bf*)(Whh + (size_t)(n0 + 32 + arow) * KP_ + k0 + koff);
            acc0 = __builtin_amdgcn_wmma_f32_16x16x32_bf16(false, a, false, b0,
                                                           (short)0, acc0, false, false);
            acc1 = __builtin_amdgcn_wmma_f32_16x16x32_bf16(false, a, false, b1,
                                                           (short)0, acc1, false, false);
            acc2 = __builtin_amdgcn_wmma_f32_16x16x32_bf16(false, a, false, b2,
                                                           (short)0, acc2, false, false);
        }
        for (int r = 0; r < 8; ++r) {
            gh[(crow + r) * H3_ + n0 +      ccol] = acc0[r];
            gh[(crow + r) * H3_ + n0 + 16 + ccol] = acc1[r];
            gh[(crow + r) * H3_ + n0 + 32 + ccol] = acc2[r];
        }
        __threadfence_block();
        __syncthreads();
        // --- GRU elementwise ---
        for (int i = threadIdx.x; i < 16 * H_; i += blockDim.x) {
            int n = i / H_, k = i % H_;
            size_t gir = ((size_t)tx * B_ + n) * H3_;
            float ir = gi[gir + k]          + bih[k];
            float iz = gi[gir + H_ + k]     + bih[H_ + k];
            float ig = gi[gir + 2 * H_ + k] + bih[2 * H_ + k];
            float hr = gh[n * H3_ + k]          + bhh[k];
            float hz = gh[n * H3_ + H_ + k]     + bhh[H_ + k];
            float hg = gh[n * H3_ + 2 * H_ + k] + bhh[2 * H_ + k];
            float r  = sigmoidf_(ir + hr);
            float z  = sigmoidf_(iz + hz);
            float g  = tanhf(ig + r * hg);
            float h2 = (1.f - z) * g + z * hcur[i];
            hcur[i] = h2;
            hbf[n * KP_ + k] = (bf16)h2;
            outp[((size_t)tx * B_ + n) * H_ + k] = h2;
        }
        __threadfence_block();
        __syncthreads();
    }
    for (int i = threadIdx.x; i < 16 * H_; i += blockDim.x) hfin[i] = hcur[i];
}

// ---------------------------------------------------------------------------
// enc_out[b][t][:] = out_f[t][b] + out_b[t][b];  enc_hidden = h_f + h_b
// ---------------------------------------------------------------------------
__global__ void k_combine_enc(const float* __restrict__ out_f,
                              const float* __restrict__ out_b,
                              const float* __restrict__ h_f,
                              const float* __restrict__ h_b,
                              float* __restrict__ enc_out,
                              float* __restrict__ enc_hid) {
    int total = B_ * TENC_ * H_;
    for (int i = blockIdx.x * blockDim.x + threadIdx.x; i < total;
         i += gridDim.x * blockDim.x) {
        int b = i / (TENC_ * H_);
        int rem = i % (TENC_ * H_);
        int t = rem / H_, k = rem % H_;
        size_t src = ((size_t)t * B_ + b) * H_ + k;
        enc_out[i] = out_f[src] + out_b[src];
    }
    for (int i = blockIdx.x * blockDim.x + threadIdx.x; i < B_ * H_;
         i += gridDim.x * blockDim.x)
        enc_hid[i] = h_f[i] + h_b[i];
}

// ---------------------------------------------------------------------------
// Decoder init: h[n] = enc_hidden[b], dec_in[n] = slot_emb[s]; bf16 copies
// (zero-pads the KP_ tail, which later steps leave untouched).
// ---------------------------------------------------------------------------
__global__ void k_dec_init(const float* __restrict__ enc_hid,
                           const float* __restrict__ slot_emb,
                           float* __restrict__ h, bf16* __restrict__ hbf,
                           float* __restrict__ din, bf16* __restrict__ dinbf) {
    int total = SB_ * KP_;
    for (int i = blockIdx.x * blockDim.x + threadIdx.x; i < total;
         i += gridDim.x * blockDim.x) {
        int n = i / KP_, c = i % KP_;
        int s = n / B_, b = n % B_;
        float hv = (c < H_) ? enc_hid[b * H_ + c] : 0.f;
        float dv = (c < H_) ? slot_emb[s * H_ + c] : 0.f;
        if (c < H_) { h[(size_t)n * H_ + c] = hv; din[(size_t)n * H_ + c] = dv; }
        hbf[i]   = (bf16)hv;
        dinbf[i] = (bf16)dv;
    }
}

// ---------------------------------------------------------------------------
// Decoder GRU elementwise combine; h updated in place + bf16 copy.
// ---------------------------------------------------------------------------
__global__ void k_dec_gru(const float* __restrict__ gi, const float* __restrict__ gh,
                          const float* __restrict__ bih, const float* __restrict__ bhh,
                          float* __restrict__ h, bf16* __restrict__ hbf) {
    int total = SB_ * H_;
    for (int i = blockIdx.x * blockDim.x + threadIdx.x; i < total;
         i += gridDim.x * blockDim.x) {
        int n = i / H_, k = i % H_;
        size_t gr = (size_t)n * H3_;
        float ir = gi[gr + k]          + bih[k];
        float iz = gi[gr + H_ + k]     + bih[H_ + k];
        float ig = gi[gr + 2 * H_ + k] + bih[2 * H_ + k];
        float hr = gh[gr + k]          + bhh[k];
        float hz = gh[gr + H_ + k]     + bhh[H_ + k];
        float hg = gh[gr + 2 * H_ + k] + bhh[2 * H_ + k];
        float r  = sigmoidf_(ir + hr);
        float z  = sigmoidf_(iz + hz);
        float g  = tanhf(ig + r * hg);
        float h2 = (1.f - z) * g + z * h[i];
        h[i] = h2;
        hbf[(size_t)n * KP_ + k] = (bf16)h2;
    }
}

// ---------------------------------------------------------------------------
// Attention + p_gen switch + (t==0) gate. One block (256 thr) per row n.
// ---------------------------------------------------------------------------
__global__ __launch_bounds__(256)
void k_attention(const float* __restrict__ h, const float* __restrict__ enc,
                 const float* __restrict__ din,
                 const float* __restrict__ W_ratio, const float* __restrict__ b_ratio,
                 const float* __restrict__ W_gate, const float* __restrict__ b_gate,
                 float* __restrict__ prob, float* __restrict__ swv,
                 float* __restrict__ gate_out, int t) {
    const int n = blockIdx.x;
    const int b = n % B_;
    const int j = threadIdx.x;

    __shared__ float h2s[H_];
    __shared__ float sc[TENC_];
    __shared__ float red[256];
    __shared__ float ctx[H_];

    for (int k = j; k < H_; k += 256) h2s[k] = h[(size_t)n * H_ + k];
    __syncthreads();

    const float* er = enc + ((size_t)b * TENC_ + j) * H_;
    float dot = 0.f;
    for (int k = 0; k < H_; ++k) dot += h2s[k] * er[k];

    red[j] = dot;
    __syncthreads();
    for (int s = 128; s > 0; s >>= 1) {
        if (j < s) red[j] = fmaxf(red[j], red[j + s]);
        __syncthreads();
    }
    float m = red[0];
    __syncthreads();
    float e = __expf(dot - m);
    red[j] = e;
    __syncthreads();
    for (int s = 128; s > 0; s >>= 1) {
        if (j < s) red[j] += red[j + s];
        __syncthreads();
    }
    float p = e / red[0];
    sc[j] = p;
    prob[(size_t)n * TENC_ + j] = p;
    __syncthreads();

    for (int k = j; k < H_; k += 256) {
        float c = 0.f;
        for (int jj = 0; jj < TENC_; ++jj)
            c += sc[jj] * enc[((size_t)b * TENC_ + jj) * H_ + k];
        ctx[k] = c;
    }
    __syncthreads();

    // sw = sigmoid([h2, ctx, dec_in] . W_ratio + b_ratio)
    float part = 0.f;
    for (int i = j; i < H3_; i += 256) {
        float v = (i < H_) ? h2s[i]
                 : (i < 2 * H_) ? ctx[i - H_]
                 : din[(size_t)n * H_ + (i - 2 * H_)];
        part += v * W_ratio[i];
    }
    __syncthreads();
    red[j] = part;
    __syncthreads();
    for (int s = 128; s > 0; s >>= 1) {
        if (j < s) red[j] += red[j + s];
        __syncthreads();
    }
    if (j == 0) swv[n] = sigmoidf_(red[0] + b_ratio[0]);

    if (t == 0 && j < G_) {
        float g = b_gate[j];
        for (int k = 0; k < H_; ++k) g += ctx[k] * W_gate[j * H_ + k];
        gate_out[n * G_ + j] = g;
    }
}

// ---------------------------------------------------------------------------
// Final per-row output: softmax over logits, scale by sw, then scatter-add
// the pointer distribution. One block (256 thr) per row.
// ---------------------------------------------------------------------------
__global__ __launch_bounds__(256)
void k_final(const float* __restrict__ logits, const float* __restrict__ prob,
             const float* __restrict__ swv, const int* __restrict__ story,
             float* __restrict__ out, int t) {
    const int n = blockIdx.x;
    const int b = n % B_;
    const int j = threadIdx.x;
    const float* lg = logits + (size_t)n * V_;
    float* o = out + ((size_t)n * TDEC_ + t) * V_;

    __shared__ float red[256];
    float m = -1e30f;
    for (int v = j; v < V_; v += 256) m = fmaxf(m, lg[v]);
    red[j] = m;
    __syncthreads();
    for (int s = 128; s > 0; s >>= 1) {
        if (j < s) red[j] = fmaxf(red[j], red[j + s]);
        __syncthreads();
    }
    m = red[0];
    __syncthreads();
    float ssum = 0.f;
    for (int v = j; v < V_; v += 256) ssum += __expf(lg[v] - m);
    red[j] = ssum;
    __syncthreads();
    for (int s = 128; s > 0; s >>= 1) {
        if (j < s) red[j] += red[j + s];
        __syncthreads();
    }
    float sw  = swv[n];
    float inv = sw / red[0];
    for (int v = j; v < V_; v += 256) o[v] = __expf(lg[v] - m) * inv;
    __threadfence();
    __syncthreads();
    // pointer scatter: one enc position per thread
    float pp = (1.f - sw) * prob[(size_t)n * TENC_ + j];
    atomicAdd(&o[story[b * TENC_ + j]], pp);
}

// ---------------------------------------------------------------------------
// Next decoder input: dec_in[n] = emb[target[b][s][t]] (+ bf16 copy).
// ---------------------------------------------------------------------------
__global__ void k_next_input(const int* __restrict__ target,
                             const float* __restrict__ emb,
                             float* __restrict__ din, bf16* __restrict__ dinbf,
                             int t) {
    int total = SB_ * H_;
    for (int i = blockIdx.x * blockDim.x + threadIdx.x; i < total;
         i += gridDim.x * blockDim.x) {
        int n = i / H_, k = i % H_;
        int s = n / B_, b = n % B_;
        int tok = target[(b * S_ + s) * TDEC_ + t];
        float v = emb[(size_t)tok * H_ + k];
        din[i] = v;
        dinbf[(size_t)n * KP_ + k] = (bf16)v;
    }
}

// ---------------------------------------------------------------------------
extern "C" void kernel_launch(void* const* d_in, const int* in_sizes, int n_in,
                              void* d_out, int out_size, void* d_ws, size_t ws_size,
                              hipStream_t stream) {
    (void)in_sizes; (void)n_in; (void)out_size; (void)ws_size;
    const int*   story   = (const int*)d_in[0];
    const int*   target  = (const int*)d_in[1];
    const float* emb     = (const float*)d_in[2];
    const float* Wih_f   = (const float*)d_in[3];
    const float* Whh_f   = (const float*)d_in[4];
    const float* bih_f   = (const float*)d_in[5];
    const float* bhh_f   = (const float*)d_in[6];
    const float* Wih_b   = (const float*)d_in[7];
    const float* Whh_b   = (const float*)d_in[8];
    const float* bih_b   = (const float*)d_in[9];
    const float* bhh_b   = (const float*)d_in[10];
    const float* dWih    = (const float*)d_in[11];
    const float* dWhh    = (const float*)d_in[12];
    const float* dbih    = (const float*)d_in[13];
    const float* dbhh    = (const float*)d_in[14];
    const float* W_ratio = (const float*)d_in[15];
    const float* b_ratio = (const float*)d_in[16];
    const float* W_gate  = (const float*)d_in[17];
    const float* b_gate  = (const float*)d_in[18];
    const float* slot_e  = (const float*)d_in[19];
    float* out = (float*)d_out;

    char* ws = (char*)d_ws;
    size_t off = 0;
    auto alloc = [&](size_t bytes) -> void* {
        void* p = ws + off;
        off = (off + bytes + 255) & ~(size_t)255;
        return p;
    };
    bf16*  emb_bf   = (bf16*) alloc((size_t)V_ * KP_ * 2);
    bf16*  x_bf     = (bf16*) alloc((size_t)TENC_ * B_ * KP_ * 2);
    bf16*  wihf_bf  = (bf16*) alloc((size_t)H3_ * KP_ * 2);
    bf16*  whhf_bf  = (bf16*) alloc((size_t)H3_ * KP_ * 2);
    bf16*  wihb_bf  = (bf16*) alloc((size_t)H3_ * KP_ * 2);
    bf16*  whhb_bf  = (bf16*) alloc((size_t)H3_ * KP_ * 2);
    bf16*  dwih_bf  = (bf16*) alloc((size_t)H3_ * KP_ * 2);
    bf16*  dwhh_bf  = (bf16*) alloc((size_t)H3_ * KP_ * 2);
    float* gi_f     = (float*)alloc((size_t)TENC_ * B_ * H3_ * 4);
    float* gi_b     = (float*)alloc((size_t)TENC_ * B_ * H3_ * 4);
    float* gh_enc   = (float*)alloc((size_t)2 * 16 * H3_ * 4);
    float* out_f    = (float*)alloc((size_t)TENC_ * B_ * H_ * 4);
    float* out_b    = (float*)alloc((size_t)TENC_ * B_ * H_ * 4);
    float* h_f      = (float*)alloc((size_t)B_ * H_ * 4);
    float* h_b      = (float*)alloc((size_t)B_ * H_ * 4);
    float* enc_out  = (float*)alloc((size_t)B_ * TENC_ * H_ * 4);
    float* enc_hid  = (float*)alloc((size_t)B_ * H_ * 4);
    float* hdec     = (float*)alloc((size_t)SB_ * H_ * 4);
    bf16*  hdec_bf  = (bf16*) alloc((size_t)SB_ * KP_ * 2);
    float* din      = (float*)alloc((size_t)SB_ * H_ * 4);
    bf16*  din_bf   = (bf16*) alloc((size_t)SB_ * KP_ * 2);
    float* gi_d     = (float*)alloc((size_t)SB_ * H3_ * 4);
    float* gh_d     = (float*)alloc((size_t)SB_ * H3_ * 4);
    float* prob     = (float*)alloc((size_t)SB_ * TENC_ * 4);
    float* swv      = (float*)alloc((size_t)SB_ * 4);
    float* logits   = (float*)alloc((size_t)SB_ * V_ * 4);

    auto gemm = [&](const bf16* A, const bf16* Bm, float* C, int M, int N) {
        int tiles  = (M / 16) * (N / 48);
        int blocks = (tiles + 7) / 8;
        k_gemm_nt<<<blocks, 256, 0, stream>>>(A, Bm, C, M, N);
    };

    // --- precision conversion / packing -----------------------------------
    k_pad_bf16<<<2048, 256, 0, stream>>>(emb,   emb_bf,  V_,  H_);
    k_pad_bf16<<<512,  256, 0, stream>>>(Wih_f, wihf_bf, H3_, H_);
    k_pad_bf16<<<512,  256, 0, stream>>>(Whh_f, whhf_bf, H3_, H_);
    k_pad_bf16<<<512,  256, 0, stream>>>(Wih_b, wihb_bf, H3_, H_);
    k_pad_bf16<<<512,  256, 0, stream>>>(Whh_b, whhb_bf, H3_, H_);
    k_pad_bf16<<<512,  256, 0, stream>>>(dWih,  dwih_bf, H3_, H_);
    k_pad_bf16<<<512,  256, 0, stream>>>(dWhh,  dwhh_bf, H3_, H_);
    k_embed_story<<<512, 256, 0, stream>>>(story, emb, x_bf);

    // --- encoder: hoisted input GEMMs + persistent recurrence -------------
    gemm(x_bf, wihf_bf, gi_f, TENC_ * B_, H3_);
    gemm(x_bf, wihb_bf, gi_b, TENC_ * B_, H3_);
    k_enc_gru<<<2, 800, 0, stream>>>(whhf_bf, whhb_bf, gi_f, gi_b,
                                     bih_f, bhh_f, bih_b, bhh_b,
                                     out_f, out_b, h_f, h_b, gh_enc);
    k_combine_enc<<<1024, 256, 0, stream>>>(out_f, out_b, h_f, h_b,
                                            enc_out, enc_hid);
    k_dec_init<<<256, 256, 0, stream>>>(enc_hid, slot_e, hdec, hdec_bf,
                                        din, din_bf);

    // --- decoder loop ------------------------------------------------------
    float* gate_base = out + (size_t)SB_ * TDEC_ * V_;
    for (int t = 0; t < TDEC_; ++t) {
        gemm(din_bf,  dwih_bf, gi_d, SB_, H3_);
        gemm(hdec_bf, dwhh_bf, gh_d, SB_, H3_);
        k_dec_gru<<<512, 256, 0, stream>>>(gi_d, gh_d, dbih, dbhh,
                                           hdec, hdec_bf);
        k_attention<<<SB_, 256, 0, stream>>>(hdec, enc_out, din,
                                             W_ratio, b_ratio, W_gate, b_gate,
                                             prob, swv, gate_base, t);
        gemm(hdec_bf, emb_bf, logits, SB_, V_);
        k_final<<<SB_, 256, 0, stream>>>(logits, prob, swv, story, out, t);
        if (t + 1 < TDEC_)
            k_next_input<<<512, 256, 0, stream>>>(target, emb, din, din_bf, t);
    }
}
